// GMNConv_2783138808164
// MI455X (gfx1250) — matrix operand order repaired and checked by reading, not compiled, not run on hardware
//
#include <hip/hip_runtime.h>

#define NN 8192      // nodes per side
#define DD 128       // feature dim
#define EE 131072    // edges per side
#define NPG 128      // nodes per graph (N/B)
#define NG 64        // graphs

typedef __attribute__((ext_vector_type(16))) _Float16 v16h;
typedef __attribute__((ext_vector_type(8)))  _Float16 v8h;
typedef __attribute__((ext_vector_type(8)))  float    v8f;

// ---------------- WMMA fragment helpers (layouts per CDNA5 ISA) ---------

__device__ __forceinline__ v16h frag_pack(v8h lo, v8h hi) {
    v16h r;
#pragma unroll
    for (int i = 0; i < 8; ++i) { r[i] = lo[i]; r[i + 8] = hi[i]; }
    return r;
}

// A fragment: 16x32 tile, row-major M x K source, ld = K stride in halves.
// lane L: row M = L%16; halves K = (L/16)*8 + [0..8) and 16+(L/16)*8 + [0..8)
__device__ __forceinline__ v16h load_frag_a(const _Float16* tile, int ld) {
    int lane = threadIdx.x & 31;
    const _Float16* p = tile + (lane & 15) * ld + ((lane >> 4) << 3);
    return frag_pack(*(const v8h*)p, *(const v8h*)(p + 16));
}

// A fragment with a per-lane row pointer (edge gather); lane's row already chosen.
__device__ __forceinline__ v16h load_frag_a_rowptr(const _Float16* rowp) {
    const _Float16* p = rowp + (((threadIdx.x >> 4) & 1) << 3);
    return frag_pack(*(const v8h*)p, *(const v8h*)(p + 16));
}

// B fragment from B^T stored N x K row-major (16x32 tile), ld = K stride in halves.
// lane L: row N = L%16; halves K = (L/16)*8 + [0..8) and 16+(L/16)*8 + [0..8)
__device__ __forceinline__ v16h load_frag_bt(const _Float16* tile, int ld) {
    int lane = threadIdx.x & 31;
    const _Float16* p = tile + (lane & 15) * ld + ((lane >> 4) << 3);
    return frag_pack(*(const v8h*)p, *(const v8h*)(p + 16));
}

__device__ __forceinline__ v8f wmma16(v16h a, v16h b, v8f c) {
    return __builtin_amdgcn_wmma_f32_16x16x32_f16(false, a, false, b, (short)0, c,
                                                  false, false);
}

__device__ __forceinline__ v8f bias_acc(float b) {
    v8f c;
#pragma unroll
    for (int j = 0; j < 8; ++j) c[j] = b;
    return c;
}

// ---------------- prep kernels ------------------------------------------------------

__global__ void transpose_w_kernel(const float* __restrict__ W, _Float16* __restrict__ WT,
                                   int fin, int fout) {
    int idx = blockIdx.x * blockDim.x + threadIdx.x;
    if (idx >= fin * fout) return;
    int i = idx / fout, o = idx % fout;
    WT[(size_t)o * fin + i] = (_Float16)W[idx];
}

// f32 x -> f16 copy, and fill the "x" slice of the update input u_h[:, 384:512]
__global__ void prep_nodes_kernel(const float* __restrict__ x, _Float16* __restrict__ xh,
                                  _Float16* __restrict__ uh) {
    int idx = blockIdx.x * blockDim.x + threadIdx.x;
    if (idx >= NN * DD) return;
    int i = idx >> 7, k = idx & 127;
    _Float16 h = (_Float16)x[idx];
    xh[idx] = h;
    uh[(size_t)i * 512 + 384 + k] = h;
}

__global__ void zero_f32_kernel(float* __restrict__ p, int n) {
    int idx = blockIdx.x * blockDim.x + threadIdx.x;
    if (idx < n) p[idx] = 0.f;
}

// messages f32 -> f16 into u_h[:, 0:256]
__global__ void msg_to_uh_kernel(const float* __restrict__ m, _Float16* __restrict__ uh) {
    int idx = blockIdx.x * blockDim.x + threadIdx.x;
    if (idx >= NN * 256) return;
    int i = idx >> 8, k = idx & 255;
    uh[(size_t)i * 512 + k] = (_Float16)m[idx];
}

// ---------------- message MLP + segment-sum scatter ---------------------------------
// one wave = one 16-edge tile; blockDim=128 (4 waves)
__global__ void __launch_bounds__(128)
message_kernel(const _Float16* __restrict__ xh, const int* __restrict__ ei,
               const _Float16* __restrict__ W1T, const float* __restrict__ b1,
               const _Float16* __restrict__ W2T, const float* __restrict__ b2,
               float* __restrict__ msum) {
    __shared__ __align__(16) _Float16 H[4][16 * 512];   // relu hidden, per wave
    __shared__ int tg[4][16];
    int w = threadIdx.x >> 5, lane = threadIdx.x & 31;
    int e0 = (blockIdx.x * 4 + w) * 16;
    int mrow = lane & 15, n = lane & 15, hh = lane >> 4;
    int tgt = ei[e0 + mrow];          // edge_index[0] = target
    int src = ei[EE + e0 + mrow];     // edge_index[1] = source
    if (lane < 16) tg[w][lane] = tgt;

    // hoist layer-1 A fragments: inp = [x[src] | x[tgt]], K tiles 0..3 src, 4..7 tgt
    v16h afr[8];
#pragma unroll
    for (int kt = 0; kt < 8; ++kt) {
        int node = (kt < 4) ? src : tgt;
        afr[kt] = load_frag_a_rowptr(xh + (size_t)node * DD + (kt & 3) * 32);
    }

    // layer 1: [16x256] @ [256x512] -> relu -> LDS f16
#pragma unroll 1
    for (int nt = 0; nt < 32; ++nt) {
        v8f c = bias_acc(b1[nt * 16 + n]);
#pragma unroll
        for (int kt = 0; kt < 8; ++kt)
            c = wmma16(afr[kt], load_frag_bt(W1T + (size_t)(nt * 16) * 256 + kt * 32, 256), c);
#pragma unroll
        for (int j = 0; j < 8; ++j) {
            float v = c[j] > 0.f ? c[j] : 0.f;
            H[w][(hh * 8 + j) * 512 + nt * 16 + n] = (_Float16)v;
        }
    }

    // layer 2: [16x512] @ [512x256] -> atomic scatter to msum[tgt]
#pragma unroll 1
    for (int nt = 0; nt < 16; ++nt) {
        v8f c = bias_acc(b2[nt * 16 + n]);
#pragma unroll
        for (int kt = 0; kt < 16; ++kt)
            c = wmma16(load_frag_a(&H[w][kt * 32], 512),
                       load_frag_bt(W2T + (size_t)(nt * 16) * 512 + kt * 32, 512), c);
#pragma unroll
        for (int j = 0; j < 8; ++j) {
            int t = tg[w][hh * 8 + j];
            atomicAdd(&msum[(size_t)t * 256 + nt * 16 + n], c[j]);
        }
    }
}

// ---------------- block-diagonal cross attention ------------------------------------
// one workgroup (8 waves) per graph; 128x128 block
__global__ void __launch_bounds__(256)
attention_kernel(const _Float16* __restrict__ x1h, const _Float16* __restrict__ x2h,
                 const float* __restrict__ x1, const float* __restrict__ x2,
                 _Float16* __restrict__ u1h, _Float16* __restrict__ u2h) {
    __shared__ __align__(16) float    S[NPG * NPG];     // 64 KB
    __shared__ __align__(16) _Float16 Ph[NPG * NPG];    // 32 KB row-softmax probs
    __shared__ __align__(16) _Float16 PcT[NPG * NPG];   // 32 KB col-softmax probs, [j][i]
    _Float16* X1T = (_Float16*)S;                       // reuse S after softmax
    _Float16* X2T = ((_Float16*)S) + NPG * NPG;

    int g = blockIdx.x;
    const _Float16* x1g = x1h + (size_t)g * NPG * DD;
    const _Float16* x2g = x2h + (size_t)g * NPG * DD;
    int w = threadIdx.x >> 5, lane = threadIdx.x & 31;
    int n = lane & 15, hh = lane >> 4;
    int m0 = w * 16;

    // phase 1: S = X1 @ X2^T  (B^T == x2 row-major, direct)
    v16h afr[4];
#pragma unroll
    for (int kt = 0; kt < 4; ++kt) afr[kt] = load_frag_a(x1g + m0 * DD + kt * 32, DD);
#pragma unroll 1
    for (int nt = 0; nt < 8; ++nt) {
        v8f c = bias_acc(0.f);
#pragma unroll
        for (int kt = 0; kt < 4; ++kt)
            c = wmma16(afr[kt], load_frag_bt(x2g + nt * 16 * DD + kt * 32, DD), c);
#pragma unroll
        for (int j = 0; j < 8; ++j)
            S[(m0 + hh * 8 + j) * NPG + nt * 16 + n] = c[j];
    }
    __syncthreads();

    // phase 2: softmax (threads 0..127: rows; 128..255: columns)
    {
        int t = threadIdx.x;
        if (t < NPG) {
            int r = t;
            float mx = -3.4e38f;
            for (int j = 0; j < NPG; ++j) mx = fmaxf(mx, S[r * NPG + j]);
            float s = 0.f;
            for (int j = 0; j < NPG; ++j) s += __expf(S[r * NPG + j] - mx);
            float inv = 1.f / s;
            for (int j = 0; j < NPG; ++j)
                Ph[r * NPG + j] = (_Float16)(__expf(S[r * NPG + j] - mx) * inv);
        } else {
            int cI = t - NPG;
            float mx = -3.4e38f;
            for (int i = 0; i < NPG; ++i) mx = fmaxf(mx, S[i * NPG + cI]);
            float s = 0.f;
            for (int i = 0; i < NPG; ++i) s += __expf(S[i * NPG + cI] - mx);
            float inv = 1.f / s;
            for (int i = 0; i < NPG; ++i)
                PcT[cI * NPG + i] = (_Float16)(__expf(S[i * NPG + cI] - mx) * inv);
        }
    }
    __syncthreads();

    // phase 3: transposed f16 copies of X1, X2 (reusing the S region)
    for (int idx = threadIdx.x; idx < NPG * DD; idx += 256) {
        int dd = idx >> 7, jj = idx & 127;
        X1T[idx] = x1g[jj * DD + dd];
        X2T[idx] = x2g[jj * DD + dd];
    }
    __syncthreads();

    // phase 4: att1 = x1 - P @ X2 -> u1h[:, 256:384]
    v16h pfr[4];
#pragma unroll
    for (int kt = 0; kt < 4; ++kt) pfr[kt] = load_frag_a(Ph + m0 * NPG + kt * 32, NPG);
#pragma unroll 1
    for (int dt = 0; dt < 8; ++dt) {
        v8f c = bias_acc(0.f);
#pragma unroll
        for (int kt = 0; kt < 4; ++kt)
            c = wmma16(pfr[kt], load_frag_bt(X2T + dt * 16 * NPG + kt * 32, NPG), c);
#pragma unroll
        for (int j = 0; j < 8; ++j) {
            int row = g * NPG + m0 + hh * 8 + j, col = dt * 16 + n;
            float v = x1[(size_t)row * DD + col] - c[j];
            u1h[(size_t)row * 512 + 256 + col] = (_Float16)v;
        }
    }

    // phase 5: att2 = x2 - Pc^T @ X1 -> u2h[:, 256:384]
#pragma unroll
    for (int kt = 0; kt < 4; ++kt) pfr[kt] = load_frag_a(PcT + m0 * NPG + kt * 32, NPG);
#pragma unroll 1
    for (int dt = 0; dt < 8; ++dt) {
        v8f c = bias_acc(0.f);
#pragma unroll
        for (int kt = 0; kt < 4; ++kt)
            c = wmma16(pfr[kt], load_frag_bt(X1T + dt * 16 * NPG + kt * 32, NPG), c);
#pragma unroll
        for (int j = 0; j < 8; ++j) {
            int row = g * NPG + m0 + hh * 8 + j, col = dt * 16 + n;
            float v = x2[(size_t)row * DD + col] - c[j];
            u2h[(size_t)row * 512 + 256 + col] = (_Float16)v;
        }
    }
}

// ---------------- node update MLP (512->512->256->128) + residual -------------------
// one wave = one 16-node tile; blockDim=128 (4 waves)
__global__ void __launch_bounds__(128)
update_kernel(const _Float16* __restrict__ uh, const float* __restrict__ x,
              const _Float16* __restrict__ W1T, const float* __restrict__ b1,
              const _Float16* __restrict__ W2T, const float* __restrict__ b2,
              const _Float16* __restrict__ W3T, const float* __restrict__ b3,
              float* __restrict__ out) {
    __shared__ __align__(16) _Float16 H1[4][16 * 512];  // 64 KB
    __shared__ __align__(16) _Float16 H2[4][16 * 256];  // 32 KB
    int w = threadIdx.x >> 5, lane = threadIdx.x & 31;
    int r0 = (blockIdx.x * 4 + w) * 16;
    int n = lane & 15, hh = lane >> 4;
    const _Float16* A = uh + (size_t)r0 * 512;

#pragma unroll 1
    for (int nt = 0; nt < 32; ++nt) {
        v8f c = bias_acc(b1[nt * 16 + n]);
#pragma unroll
        for (int kt = 0; kt < 16; ++kt)
            c = wmma16(load_frag_a(A + kt * 32, 512),
                       load_frag_bt(W1T + (size_t)(nt * 16) * 512 + kt * 32, 512), c);
#pragma unroll
        for (int j = 0; j < 8; ++j) {
            float v = c[j] > 0.f ? c[j] : 0.f;
            H1[w][(hh * 8 + j) * 512 + nt * 16 + n] = (_Float16)v;
        }
    }
#pragma unroll 1
    for (int nt = 0; nt < 16; ++nt) {
        v8f c = bias_acc(b2[nt * 16 + n]);
#pragma unroll
        for (int kt = 0; kt < 16; ++kt)
            c = wmma16(load_frag_a(&H1[w][kt * 32], 512),
                       load_frag_bt(W2T + (size_t)(nt * 16) * 512 + kt * 32, 512), c);
#pragma unroll
        for (int j = 0; j < 8; ++j) {
            float v = c[j] > 0.f ? c[j] : 0.f;
            H2[w][(hh * 8 + j) * 256 + nt * 16 + n] = (_Float16)v;
        }
    }
#pragma unroll 1
    for (int nt = 0; nt < 8; ++nt) {
        v8f c = bias_acc(b3[nt * 16 + n]);
#pragma unroll
        for (int kt = 0; kt < 8; ++kt)
            c = wmma16(load_frag_a(&H2[w][kt * 32], 256),
                       load_frag_bt(W3T + (size_t)(nt * 16) * 256 + kt * 32, 256), c);
#pragma unroll
        for (int j = 0; j < 8; ++j) {
            int row = r0 + hh * 8 + j, col = nt * 16 + n;
            out[(size_t)row * DD + col] = x[(size_t)row * DD + col] + c[j];
        }
    }
}

// ---------------- host launch -------------------------------------------------------

extern "C" void kernel_launch(void* const* d_in, const int* in_sizes, int n_in,
                              void* d_out, int out_size, void* d_ws, size_t ws_size,
                              hipStream_t stream) {
    const float* x1  = (const float*)d_in[0];
    const int*   ei1 = (const int*)d_in[1];
    const float* x2  = (const float*)d_in[3];
    const int*   ei2 = (const int*)d_in[4];
    const float* mW1 = (const float*)d_in[6];
    const float* mb1 = (const float*)d_in[7];
    const float* mW2 = (const float*)d_in[8];
    const float* mb2 = (const float*)d_in[9];
    const float* uW1 = (const float*)d_in[10];
    const float* ub1 = (const float*)d_in[11];
    const float* uW2 = (const float*)d_in[12];
    const float* ub2 = (const float*)d_in[13];
    const float* uW3 = (const float*)d_in[14];
    const float* ub3 = (const float*)d_in[15];
    float* out = (float*)d_out;

    char* ws = (char*)d_ws;
    size_t off = 0;
    auto alloc = [&](size_t bytes) -> void* {
        void* p = ws + off;
        off = (off + bytes + 255) & ~(size_t)255;
        return p;
    };
    _Float16* x1h  = (_Float16*)alloc((size_t)NN * DD * 2);
    _Float16* x2h  = (_Float16*)alloc((size_t)NN * DD * 2);
    _Float16* mW1T = (_Float16*)alloc((size_t)512 * 256 * 2);
    _Float16* mW2T = (_Float16*)alloc((size_t)256 * 512 * 2);
    _Float16* uW1T = (_Float16*)alloc((size_t)512 * 512 * 2);
    _Float16* uW2T = (_Float16*)alloc((size_t)256 * 512 * 2);
    _Float16* uW3T = (_Float16*)alloc((size_t)128 * 256 * 2);
    float*    m1   = (float*)alloc((size_t)NN * 256 * 4);
    float*    m2   = (float*)alloc((size_t)NN * 256 * 4);
    _Float16* u1h  = (_Float16*)alloc((size_t)NN * 512 * 2);
    _Float16* u2h  = (_Float16*)alloc((size_t)NN * 512 * 2);

    // weight transpose + f16 convert
    transpose_w_kernel<<<(256 * 512 + 255) / 256, 256, 0, stream>>>(mW1, mW1T, 256, 512);
    transpose_w_kernel<<<(512 * 256 + 255) / 256, 256, 0, stream>>>(mW2, mW2T, 512, 256);
    transpose_w_kernel<<<(512 * 512 + 255) / 256, 256, 0, stream>>>(uW1, uW1T, 512, 512);
    transpose_w_kernel<<<(512 * 256 + 255) / 256, 256, 0, stream>>>(uW2, uW2T, 512, 256);
    transpose_w_kernel<<<(256 * 128 + 255) / 256, 256, 0, stream>>>(uW3, uW3T, 256, 128);

    // node feature f16 copies + u_h x-slice
    prep_nodes_kernel<<<(NN * DD + 255) / 256, 256, 0, stream>>>(x1, x1h, u1h);
    prep_nodes_kernel<<<(NN * DD + 255) / 256, 256, 0, stream>>>(x2, x2h, u2h);

    // zero message accumulators (atomics)
    zero_f32_kernel<<<(NN * 256 + 255) / 256, 256, 0, stream>>>(m1, NN * 256);
    zero_f32_kernel<<<(NN * 256 + 255) / 256, 256, 0, stream>>>(m2, NN * 256);

    // message MLP + scatter (both sides)
    message_kernel<<<EE / 16 / 4, 128, 0, stream>>>(x1h, ei1, mW1T, mb1, mW2T, mb2, m1);
    message_kernel<<<EE / 16 / 4, 128, 0, stream>>>(x2h, ei2, mW1T, mb1, mW2T, mb2, m2);

    // block-diagonal cross attention (writes att slices of u1h/u2h)
    attention_kernel<<<NG, 256, 0, stream>>>(x1h, x2h, x1, x2, u1h, u2h);

    // message slices of u_h
    msg_to_uh_kernel<<<(NN * 256 + 255) / 256, 256, 0, stream>>>(m1, u1h);
    msg_to_uh_kernel<<<(NN * 256 + 255) / 256, 256, 0, stream>>>(m2, u2h);

    // node update MLP + residual -> d_out (new_x1 then new_x2)
    update_kernel<<<NN / 16 / 4, 128, 0, stream>>>(u1h, x1, uW1T, ub1, uW2T, ub2, uW3T,
                                                   ub3, out);
    update_kernel<<<NN / 16 / 4, 128, 0, stream>>>(u2h, x2, uW1T, ub1, uW2T, ub2, uW3T,
                                                   ub3, out + (size_t)NN * DD);
}
